// ActionEncoder_53970559042346
// MI455X (gfx1250) — matrix operand order repaired and checked by reading, not compiled, not run on hardware
//
#include <hip/hip_runtime.h>

typedef __attribute__((ext_vector_type(16))) _Float16 v16h_t;
typedef __attribute__((ext_vector_type(8)))  float    v8f_t;

#define N_PRED  2550
#define MAX_N   50
#define BATCH   16384
#define KP      2560            // K padded to multiple of 32
#define NP      2560            // N padded to multiple of 16
#define KTILES  (KP / 32)       // 80
#define NTILES  (NP / 16)       // 160
#define M0ROWS  64              // expert-0 table rows padded (50 -> 64)
#define M1ROWS  2528            // expert-1 table rows padded (2500 -> 2528, even #tiles)
#define M0TILES (M0ROWS / 16)   // 4
#define M1TILES (M1ROWS / 16)   // 158

// ---------------------------------------------------------------------------
// Pack W2 (f32, [q, p] row-major) into the per-lane WMMA B-operand layout
// (f16).  B-matrix 32x16 (16-bit): lane = hi*16 + n, element e -> K = (e&7) + hi*8 + (e>>3)*16.
// Layout: Wp[((nt*KTILES + kt)*32 + lane)*16 + e]
// ---------------------------------------------------------------------------
__global__ void pack_w2_kernel(const float* __restrict__ W2,
                               _Float16* __restrict__ Wp) {
    size_t t = (size_t)blockIdx.x * blockDim.x + threadIdx.x;   // exact size
    int e  = (int)(t & 15);
    int L  = (int)((t >> 4) & 31);
    int kt = (int)((t >> 9) % KTILES);
    int nt = (int)((t >> 9) / KTILES);
    int n  = nt * 16 + (L & 15);
    int k  = kt * 32 + (e & 7) + (L >> 4) * 8 + (e >> 3) * 16;
    float v = 0.0f;
    if (n < N_PRED && k < N_PRED) v = W2[(size_t)n * N_PRED + k];
    Wp[t] = (_Float16)v;
}

// A-operand K interleaving (16-bit A 16x32, ISA 7.12.2):
// lane = hi*16 + m, element e (j=e>>1, par=e&1):
//   k = (j&3)*2 + par + ((j>>2)<<4) + hi*8
__device__ __forceinline__ int a_kmap(int e, int hi) {
    int j = e >> 1, par = e & 1;
    return (j & 3) * 2 + par + ((j >> 2) << 4) + hi * 8;
}

// Expert 0: H0[m, k] = relu(W1_0[k*50 + m] + b1_0[k]),  m<50, k<2550
__global__ void pack_h0_kernel(const float* __restrict__ W1,
                               const float* __restrict__ b1,
                               _Float16* __restrict__ Hp) {
    size_t t = (size_t)blockIdx.x * blockDim.x + threadIdx.x;
    int e  = (int)(t & 15);
    int L  = (int)((t >> 4) & 31);
    int kt = (int)((t >> 9) % KTILES);
    int mt = (int)((t >> 9) / KTILES);
    int hi = L >> 4;
    int m  = mt * 16 + (L & 15);
    int k  = kt * 32 + a_kmap(e, hi);
    float v = 0.0f;
    if (m < MAX_N && k < N_PRED) {
        v = W1[(size_t)k * MAX_N + m] + b1[k];
        v = v > 0.0f ? v : 0.0f;
    }
    Hp[t] = (_Float16)v;
}

// Expert 1: row r = i0*50 + i1:
//   H1[r, k] = relu(W1_1[k*100 + i0] + W1_1[k*100 + 50 + i1] + b1_1[k])
__global__ void pack_h1_kernel(const float* __restrict__ W1,
                               const float* __restrict__ b1,
                               _Float16* __restrict__ Hp) {
    size_t t = (size_t)blockIdx.x * blockDim.x + threadIdx.x;
    int e  = (int)(t & 15);
    int L  = (int)((t >> 4) & 31);
    int kt = (int)((t >> 9) % KTILES);
    int mt = (int)((t >> 9) / KTILES);
    int hi = L >> 4;
    int r  = mt * 16 + (L & 15);
    int k  = kt * 32 + a_kmap(e, hi);
    float v = 0.0f;
    if (r < MAX_N * MAX_N && k < N_PRED) {
        int i0 = r / MAX_N, i1 = r % MAX_N;
        v = W1[(size_t)k * (2 * MAX_N) + i0]
          + W1[(size_t)k * (2 * MAX_N) + MAX_N + i1] + b1[k];
        v = v > 0.0f ? v : 0.0f;
    }
    Hp[t] = (_Float16)v;
}

__device__ __forceinline__ float trinary(float x) {
    return x > 0.5f ? 1.0f : (x < -0.5f ? -1.0f : 0.0f);
}

// ---------------------------------------------------------------------------
// Table GEMM: T[m, n] = trinary(sum_k H[m,k] * W2[n,k] + b2[n])
// Register-blocked: each wave owns a 32x64 output tile (2 M-tiles x 4 N-tiles)
// -> per K-step: 2 A-frag + 4 B-frag loads (12x b128), 8 WMMAs.
// 8 waves/block cover 8*4 = 32 N-tiles; grid = (NTILES/32, Mtiles/2).
// ---------------------------------------------------------------------------
__global__ void gemm_trinary_kernel(const _Float16* __restrict__ Hp,
                                    const _Float16* __restrict__ Wp,
                                    const float* __restrict__ b2,
                                    float* __restrict__ T) {
    int wave = threadIdx.x >> 5;
    int lane = threadIdx.x & 31;
    int nt0  = (blockIdx.x * 8 + wave) * 4;   // first of 4 N-tiles
    int mt0  = blockIdx.y * 2;                // first of 2 M-tiles

    const size_t TILE_STRIDE = (size_t)KTILES * 32 * 16;  // halfs per operand tile
    const _Float16* ap0 = Hp + TILE_STRIDE * (mt0 + 0) + (size_t)lane * 16;
    const _Float16* ap1 = Hp + TILE_STRIDE * (mt0 + 1) + (size_t)lane * 16;
    const _Float16* bp0 = Wp + TILE_STRIDE * (nt0 + 0) + (size_t)lane * 16;
    const _Float16* bp1 = Wp + TILE_STRIDE * (nt0 + 1) + (size_t)lane * 16;
    const _Float16* bp2 = Wp + TILE_STRIDE * (nt0 + 2) + (size_t)lane * 16;
    const _Float16* bp3 = Wp + TILE_STRIDE * (nt0 + 3) + (size_t)lane * 16;

    v8f_t c00 = {}, c01 = {}, c02 = {}, c03 = {};
    v8f_t c10 = {}, c11 = {}, c12 = {}, c13 = {};

    #pragma unroll 2
    for (int kt = 0; kt < KTILES; ++kt) {
        v16h_t a0 = *(const v16h_t*)ap0;
        v16h_t a1 = *(const v16h_t*)ap1;
        v16h_t b0 = *(const v16h_t*)bp0;
        v16h_t b1 = *(const v16h_t*)bp1;
        v16h_t b2v = *(const v16h_t*)bp2;
        v16h_t b3 = *(const v16h_t*)bp3;
        c00 = __builtin_amdgcn_wmma_f32_16x16x32_f16(false, a0, false, b0,  (short)0, c00, false, false);
        c01 = __builtin_amdgcn_wmma_f32_16x16x32_f16(false, a0, false, b1,  (short)0, c01, false, false);
        c02 = __builtin_amdgcn_wmma_f32_16x16x32_f16(false, a0, false, b2v, (short)0, c02, false, false);
        c03 = __builtin_amdgcn_wmma_f32_16x16x32_f16(false, a0, false, b3,  (short)0, c03, false, false);
        c10 = __builtin_amdgcn_wmma_f32_16x16x32_f16(false, a1, false, b0,  (short)0, c10, false, false);
        c11 = __builtin_amdgcn_wmma_f32_16x16x32_f16(false, a1, false, b1,  (short)0, c11, false, false);
        c12 = __builtin_amdgcn_wmma_f32_16x16x32_f16(false, a1, false, b2v, (short)0, c12, false, false);
        c13 = __builtin_amdgcn_wmma_f32_16x16x32_f16(false, a1, false, b3,  (short)0, c13, false, false);
        ap0 += 512; ap1 += 512;                 // 32 lanes * 16 halfs per step
        bp0 += 512; bp1 += 512; bp2 += 512; bp3 += 512;
    }

    // C/D layout: VGPR r -> M = r + 8*(lane>>4), N = lane & 15
    int hi    = lane >> 4;
    int nbase = nt0 * 16 + (lane & 15);
    v8f_t acc[2][4] = {{c00, c01, c02, c03}, {c10, c11, c12, c13}};
    #pragma unroll
    for (int jm = 0; jm < 2; ++jm) {
        size_t rowBase = (size_t)((mt0 + jm) * 16 + hi * 8) * NP;
        #pragma unroll
        for (int jn = 0; jn < 4; ++jn) {
            int n = nbase + jn * 16;
            float bias = (n < N_PRED) ? b2[n] : 0.0f;
            #pragma unroll
            for (int r = 0; r < 8; ++r)
                T[rowBase + (size_t)r * NP + n] = trinary(acc[jm][jn][r] + bias);
        }
    }
}

// ---------------------------------------------------------------------------
// Per-batch-row routed table lookup.  Tables (~26 MB) stay L2-resident; this
// stage is bound by the 167 MB output write (~7 us at 23.3 TB/s).
// Row base offsets are 8-byte aligned for every b (2550*4 % 8 == 0), so copy
// as float2 (b64 loads/stores).
// ---------------------------------------------------------------------------
__global__ void gather_kernel(const int* __restrict__ idx,
                              const int* __restrict__ types,
                              const float* __restrict__ T0,
                              const float* __restrict__ T1,
                              float* __restrict__ out) {
    int b  = blockIdx.x;
    int ty = types[b];
    int i0 = idx[2 * b + 0];
    int i1 = idx[2 * b + 1];
    const float2* src = (const float2*)((ty == 0)
        ? (T0 + (size_t)i0 * NP)
        : (T1 + ((size_t)i0 * MAX_N + i1) * NP));
    float2* dst = (float2*)(out + (size_t)b * N_PRED);
    const int NV2 = N_PRED / 2;                 // 1275 float2 per row
    for (int q = threadIdx.x; q < NV2; q += blockDim.x)
        dst[q] = src[q];
}

extern "C" void kernel_launch(void* const* d_in, const int* in_sizes, int n_in,
                              void* d_out, int out_size, void* d_ws, size_t ws_size,
                              hipStream_t stream) {
    (void)in_sizes; (void)n_in; (void)out_size; (void)ws_size;

    const int*   action_indices = (const int*)d_in[0];
    // d_in[1] = action_n_obj (redundant: implied by action_types)
    const int*   action_types   = (const int*)d_in[2];
    const float* W1_0 = (const float*)d_in[3];
    const float* b1_0 = (const float*)d_in[4];
    const float* W2_0 = (const float*)d_in[5];
    const float* b2_0 = (const float*)d_in[6];
    const float* W1_1 = (const float*)d_in[7];
    const float* b1_1 = (const float*)d_in[8];
    const float* W2_1 = (const float*)d_in[9];
    const float* b2_1 = (const float*)d_in[10];

    // Workspace carve-up (all offsets 32-byte aligned): ~66 MB total.
    _Float16* W2p0 = (_Float16*)d_ws;
    _Float16* W2p1 = W2p0 + (size_t)KP * NP;
    _Float16* Hp0  = W2p1 + (size_t)KP * NP;
    _Float16* Hp1  = Hp0  + (size_t)M0ROWS * KP;
    float*    T0   = (float*)(Hp1 + (size_t)M1ROWS * KP);
    float*    T1   = T0 + (size_t)M0ROWS * NP;

    const int PACK_W2_THREADS = NTILES  * KTILES * 32 * 16;  // 6,553,600
    const int PACK_H0_THREADS = M0TILES * KTILES * 32 * 16;  //   163,840
    const int PACK_H1_THREADS = M1TILES * KTILES * 32 * 16;  // 6,471,680

    pack_w2_kernel<<<PACK_W2_THREADS / 256, 256, 0, stream>>>(W2_0, W2p0);
    pack_w2_kernel<<<PACK_W2_THREADS / 256, 256, 0, stream>>>(W2_1, W2p1);
    pack_h0_kernel<<<PACK_H0_THREADS / 256, 256, 0, stream>>>(W1_0, b1_0, Hp0);
    pack_h1_kernel<<<PACK_H1_THREADS / 256, 256, 0, stream>>>(W1_1, b1_1, Hp1);

    // Each block: 8 waves x (32x64) = 32 (M) x 512 (N); grid.x covers 2560 N.
    gemm_trinary_kernel<<<dim3(NTILES / 32, M0TILES / 2), 256, 0, stream>>>(Hp0, W2p0, b2_0, T0);
    gemm_trinary_kernel<<<dim3(NTILES / 32, M1TILES / 2), 256, 0, stream>>>(Hp1, W2p1, b2_1, T1);

    gather_kernel<<<BATCH, 256, 0, stream>>>(action_indices, action_types, T0, T1,
                                             (float*)d_out);
}
